// NTXentLoss1_40535901340191
// MI455X (gfx1250) — matrix operand order repaired and checked by reading, not compiled, not run on hardware
//
#include <hip/hip_runtime.h>
#include <hip/hip_bf16.h>

#define M_TOT   8192
#define B_HALF  4096
#define DIM     256
#define INVTEMP 10.0f
#define SHIFT   10.0f
#define EPS     1e-8f
#define RB      4                                // A row-blocks per wave
#define ROWS_PER_WAVE   (RB * 16)                // 64
#define NCSPLIT 16
#define COLS_PER_CHUNK  (M_TOT / NCSPLIT)        // 512
#define TILES_PER_CHUNK (COLS_PER_CHUNK / 16)    // 32

typedef __attribute__((ext_vector_type(16))) __bf16 v16bf;
typedef __attribute__((ext_vector_type(8)))  float  v8f;

__device__ __forceinline__ unsigned short f2bf(float f) {
    unsigned u = __float_as_uint(f);
    u += 0x7FFFu + ((u >> 16) & 1u);   // round-to-nearest-even
    return (unsigned short)(u >> 16);
}
__device__ __forceinline__ float bflo(unsigned u) { return __uint_as_float(u << 16); }
__device__ __forceinline__ float bfhi(unsigned u) { return __uint_as_float(u & 0xFFFF0000u); }

// ---------------------------------------------------------------------------
// Kernel 1: row-normalize x (f32) -> bf16 x-hat. One wave per row.
// ---------------------------------------------------------------------------
__global__ void __launch_bounds__(256) ntx_normalize(const float* __restrict__ x,
                                                     unsigned short* __restrict__ xn) {
    int wave = threadIdx.x >> 5;
    int lane = threadIdx.x & 31;
    int row  = blockIdx.x * 8 + wave;

    const float4* p = (const float4*)(x + (size_t)row * DIM + lane * 8);
    float4 a = p[0], b = p[1];
    float ss = a.x*a.x + a.y*a.y + a.z*a.z + a.w*a.w
             + b.x*b.x + b.y*b.y + b.z*b.z + b.w*b.w;
    #pragma unroll
    for (int m = 16; m >= 1; m >>= 1) ss += __shfl_xor(ss, m, 32);

    float inv = 1.0f / fmaxf(sqrtf(ss), EPS);

    uint4 o;
    o.x = (unsigned)f2bf(a.x*inv) | ((unsigned)f2bf(a.y*inv) << 16);
    o.y = (unsigned)f2bf(a.z*inv) | ((unsigned)f2bf(a.w*inv) << 16);
    o.z = (unsigned)f2bf(b.x*inv) | ((unsigned)f2bf(b.y*inv) << 16);
    o.w = (unsigned)f2bf(b.z*inv) | ((unsigned)f2bf(b.w*inv) << 16);
    *(uint4*)(xn + (size_t)row * DIM + lane * 8) = o;
}

// ---------------------------------------------------------------------------
// Kernel 2: fused S = (x^ x^T)/tau GEMM (bf16 WMMA, f32 accum) + fixed-shift
// exp row-sum over ALL columns (diagonal subtracted analytically later).
// Branchless hot loop. Each wave holds FOUR 16-row A blocks (full K=256,
// 256 VGPRs via the gfx1250 1024-VGPR mechanism) so every streamed B chunk
// feeds 4 WMMAs: 256 B of cache feed per WMMA = balanced with ~1 WMMA/cyc/WGP.
// B chunks are explicitly double-buffered: loads for chunk k+1 issue before
// the WMMA quad of chunk k.
// ---------------------------------------------------------------------------
__global__ void __launch_bounds__(256) ntx_sim(const unsigned short* __restrict__ xnu,
                                               float* __restrict__ ssum_part) {
    int wave = threadIdx.x >> 5;
    int lane = threadIdx.x & 31;
    int half = lane >> 4;          // which 16-lane group
    int lc   = lane & 15;
    int r0 = __builtin_amdgcn_readfirstlane((blockIdx.x * 8 + wave) * ROWS_PER_WAVE);
    int cc = blockIdx.y;

    // ---- Preload A fragments: RB*16 rows, all 8 K-chunks of 32 ----
    // 16-bit A 16x32 layout: lanes 0-15 row=lc K=[k0..k0+7 | k0+16..k0+23],
    //                        lanes 16-31 row=lc K=[k0+8..k0+15 | k0+24..k0+31].
    v16bf afrag[RB][8];
    #pragma unroll
    for (int rb = 0; rb < RB; ++rb) {
        const unsigned short* abase =
            xnu + (size_t)(r0 + rb * 16 + lc) * DIM + half * 8;
        #pragma unroll
        for (int kc = 0; kc < 8; ++kc) {
            union { uint4 u[2]; v16bf v; } U;
            U.u[0] = *(const uint4*)(abase + kc * 32);
            U.u[1] = *(const uint4*)(abase + kc * 32 + 16);
            afrag[rb][kc] = U.v;
        }
    }

    float ssum[RB][8];
    #pragma unroll
    for (int rb = 0; rb < RB; ++rb)
        #pragma unroll
        for (int m = 0; m < 8; ++m) ssum[rb][m] = 0.0f;

    union BU { uint4 u[2]; v16bf v; };

    for (int ct = 0; ct < TILES_PER_CHUNK; ++ct) {
        int c0 = cc * COLS_PER_CHUNK + ct * 16;
        // 16-bit B 32x16 layout: lanes 0-15 col=lc K=[k0..k0+7 | k0+16..k0+23],
        //                        lanes 16-31 col=lc K=[k0+8..k0+15 | k0+24..k0+31].
        const unsigned short* bb = xnu + (size_t)(c0 + lc) * DIM + half * 8;

        if (ct + 1 < TILES_PER_CHUNK)
            __builtin_prefetch(bb + 16 * DIM, 0, 1);   // global_prefetch_b8

        v8f acc[RB];
        #pragma unroll
        for (int rb = 0; rb < RB; ++rb)
            acc[rb] = (v8f){0.f, 0.f, 0.f, 0.f, 0.f, 0.f, 0.f, 0.f};

        // Software-pipelined B stream: 2-deep register double buffer.
        BU buf[2];
        buf[0].u[0] = *(const uint4*)(bb + 0);
        buf[0].u[1] = *(const uint4*)(bb + 16);
        #pragma unroll
        for (int kc = 0; kc < 8; ++kc) {
            if (kc + 1 < 8) {
                buf[(kc + 1) & 1].u[0] = *(const uint4*)(bb + (kc + 1) * 32);
                buf[(kc + 1) & 1].u[1] = *(const uint4*)(bb + (kc + 1) * 32 + 16);
            }
            #pragma unroll
            for (int rb = 0; rb < RB; ++rb)
                acc[rb] = __builtin_amdgcn_wmma_f32_16x16x32_bf16(
                              false, afrag[rb][kc], false, buf[kc & 1].v,
                              (short)0, acc[rb], false, false);
        }

        #pragma unroll
        for (int rb = 0; rb < RB; ++rb)
            #pragma unroll
            for (int m = 0; m < 8; ++m)
                ssum[rb][m] += __expf(fmaf(acc[rb][m], INVTEMP, -SHIFT));
    }

    // Reduce across the 16-lane group (xor 1,2,4,8 stays inside the group).
    #pragma unroll
    for (int rb = 0; rb < RB; ++rb) {
        #pragma unroll
        for (int m = 0; m < 8; ++m) {
            float s = ssum[rb][m];
            #pragma unroll
            for (int d = 1; d < 16; d <<= 1) s += __shfl_xor(s, d, 32);
            if (lc == 0) {
                int rowg = r0 + rb * 16 + m + half * 8;
                ssum_part[rowg * NCSPLIT + cc] = s;
            }
        }
    }
}

// ---------------------------------------------------------------------------
// Kernel 3: per-row epilogue. One wave per row:
//   self = dot(xn_i, xn_i)          (exact f32 products of bf16 values)
//   pos  = dot(xn_i, xn_{i^4096})   (the positive logit / INVTEMP)
//   loss_i = SHIFT - pos*INVTEMP + log( sum_part_total - exp(self*10 - 10) )
// ---------------------------------------------------------------------------
__global__ void __launch_bounds__(256) ntx_rowloss(const unsigned short* __restrict__ xnu,
                                                   const float* __restrict__ ssum_part,
                                                   float* __restrict__ rowloss) {
    int wave = threadIdx.x >> 5;
    int lane = threadIdx.x & 31;
    int row  = blockIdx.x * 8 + wave;
    int prow = row ^ B_HALF;

    uint4 ua = *(const uint4*)(xnu + (size_t)row  * DIM + lane * 8);
    uint4 ub = *(const uint4*)(xnu + (size_t)prow * DIM + lane * 8);

    float sself = 0.0f, spos = 0.0f;
    {
        float a0 = bflo(ua.x), a1 = bfhi(ua.x), a2 = bflo(ua.y), a3 = bfhi(ua.y);
        float a4 = bflo(ua.z), a5 = bfhi(ua.z), a6 = bflo(ua.w), a7 = bfhi(ua.w);
        float b0 = bflo(ub.x), b1 = bfhi(ub.x), b2 = bflo(ub.y), b3 = bfhi(ub.y);
        float b4 = bflo(ub.z), b5 = bfhi(ub.z), b6 = bflo(ub.w), b7 = bfhi(ub.w);
        sself = a0*a0 + a1*a1 + a2*a2 + a3*a3 + a4*a4 + a5*a5 + a6*a6 + a7*a7;
        spos  = a0*b0 + a1*b1 + a2*b2 + a3*b3 + a4*b4 + a5*b5 + a6*b6 + a7*b7;
    }
    #pragma unroll
    for (int d = 16; d >= 1; d >>= 1) {
        sself += __shfl_xor(sself, d, 32);
        spos  += __shfl_xor(spos,  d, 32);
    }

    if (lane == 0) {
        float s = 0.0f;
        #pragma unroll
        for (int j = 0; j < NCSPLIT; ++j) s += ssum_part[row * NCSPLIT + j];
        s -= __expf(fmaf(sself, INVTEMP, -SHIFT));      // remove diagonal term
        rowloss[row] = (SHIFT - spos * INVTEMP) + logf(s);
    }
}

// ---------------------------------------------------------------------------
// Kernel 4: deterministic fixed-order scalar reduction -> mean loss.
// ---------------------------------------------------------------------------
__global__ void __launch_bounds__(256) ntx_finalize(const float* __restrict__ rowloss,
                                                    float* __restrict__ out) {
    __shared__ float red[256];
    int t = threadIdx.x;
    float acc = 0.0f;
    for (int r = t; r < M_TOT; r += 256) acc += rowloss[r];
    red[t] = acc;
    __syncthreads();
    for (int w = 128; w > 0; w >>= 1) {
        if (t < w) red[t] += red[t + w];
        __syncthreads();
    }
    if (t == 0) out[0] = red[0] / (float)M_TOT;
}

// ---------------------------------------------------------------------------
extern "C" void kernel_launch(void* const* d_in, const int* in_sizes, int n_in,
                              void* d_out, int out_size, void* d_ws, size_t ws_size,
                              hipStream_t stream) {
    (void)in_sizes; (void)n_in; (void)out_size; (void)ws_size;

    const float* x = (const float*)d_in[0];

    // Workspace: [ xn bf16 : 4 MB ][ ssum 8192*16 f32 : 512 KB ][ rowloss : 32 KB ]
    unsigned short* xn = (unsigned short*)d_ws;
    float* ssum_part = (float*)((char*)d_ws + (size_t)M_TOT * DIM * 2);
    float* rowloss   = ssum_part + (size_t)M_TOT * NCSPLIT;

    ntx_normalize<<<M_TOT / 8, 256, 0, stream>>>(x, xn);

    // 16 row-blocks x 16 column chunks, 8 waves/block, 64 rows/wave = 2048 waves
    dim3 g2(M_TOT / (8 * ROWS_PER_WAVE), NCSPLIT);
    ntx_sim<<<g2, 256, 0, stream>>>(xn, ssum_part);

    ntx_rowloss<<<M_TOT / 8, 256, 0, stream>>>(xn, ssum_part, rowloss);
    ntx_finalize<<<1, 256, 0, stream>>>(rowloss, (float*)d_out);
}
